// Encoder_6090263626284
// MI455X (gfx1250) — matrix operand order repaired and checked by reading, not compiled, not run on hardware
//
#include <hip/hip_runtime.h>

// Problem dimensions (fixed by the reference).
#define T_DIM 512
#define B_DIM 256
#define V_DIM 128
#define E_DIM 256
#define H_DIM 512
#define G_DIM (4 * H_DIM)  // 2048

typedef __attribute__((ext_vector_type(16))) __bf16        bf16x16;
typedef __attribute__((ext_vector_type(8)))  float         f32x8;
typedef __attribute__((ext_vector_type(4)))  unsigned int  u32x4;

struct U32x8 { u32x4 lo; u32x4 hi; };

// 32 contiguous bytes (16 bf16) -> WMMA fragment register block.
__device__ __forceinline__ bf16x16 load_bf16x16(const unsigned short* p) {
  U32x8 u;
  u.lo = *(const u32x4*)(p);
  u.hi = *(const u32x4*)(p + 8);
  return __builtin_bit_cast(bf16x16, u);
}

// Two disjoint 16-byte chunks (A-matrix lane layout needs K and K+16 halves).
__device__ __forceinline__ bf16x16 load_bf16x16_split(const unsigned short* p0,
                                                      const unsigned short* p1) {
  U32x8 u;
  u.lo = *(const u32x4*)(p0);
  u.hi = *(const u32x4*)(p1);
  return __builtin_bit_cast(bf16x16, u);
}

__device__ __forceinline__ unsigned short f32_to_bf16_rne(float f) {
  unsigned int u = __builtin_bit_cast(unsigned int, f);
  u += 0x7FFFu + ((u >> 16) & 1u);  // round-to-nearest-even
  return (unsigned short)(u >> 16);
}

__device__ __forceinline__ float sigmoidf_fast(float x) {
  return 1.0f / (1.0f + __expf(-x));
}

__device__ __forceinline__ float tanhf_fast(float x) {
  float e2 = __expf(-2.0f * fabsf(x));
  float t  = (1.0f - e2) / (1.0f + e2);
  return copysignf(t, x);
}

// ---------------------------------------------------------------------------
// fp32 -> bf16 conversion (weights / embedding).
__global__ void cvt_f32_bf16_kernel(const float* __restrict__ in,
                                    unsigned short* __restrict__ out, int n) {
  int i = blockIdx.x * 256 + threadIdx.x;
  if (i < n) out[i] = f32_to_bf16_rne(in[i]);
}

// Zero-init c, h (fp32) and the bf16 hidden ping buffer.
__global__ void init_state_kernel(float* __restrict__ c, float* __restrict__ h,
                                  unsigned short* __restrict__ hb, int n) {
  int i = blockIdx.x * 256 + threadIdx.x;
  if (i < n) { c[i] = 0.0f; h[i] = 0.0f; hb[i] = 0; }
}

// ---------------------------------------------------------------------------
// G[v, g] = (emb @ W_ih^T)[v, g] + b_ih[g] + b_hh[g]   (V x 4H table, 1 MB)
// Block: 8 waves, each wave a 16x32 tile; grid (4H/256, V/16).
__global__ __launch_bounds__(256) void g_table_kernel(
    const unsigned short* __restrict__ emb_bf,   // [V][E] bf16 row-major
    const unsigned short* __restrict__ wih_bf,   // [4H][E] bf16 row-major
    const float* __restrict__ b_ih,
    const float* __restrict__ b_hh,
    float* __restrict__ Gtab) {                  // [V][4H] f32
  const int lane   = threadIdx.x & 31;
  const int wave   = threadIdx.x >> 5;
  const int row0   = blockIdx.y * 16;
  const int col0   = blockIdx.x * 256 + wave * 32;
  const int laneLo = lane & 15;
  const int laneHi = lane >> 4;

  // A: emb rows (M = vocab id), 16x32 bf16 lane layout.
  const unsigned short* a_row = emb_bf + (size_t)(row0 + laneLo) * E_DIM;
  const int a_off = laneHi * 8;
  // B: W_ih^T columns == W_ih rows, contiguous in k.
  const unsigned short* b0_row = wih_bf + (size_t)(col0 + laneLo) * E_DIM;
  const unsigned short* b1_row = b0_row + 16 * E_DIM;

  f32x8 acc0 = {};
  f32x8 acc1 = {};
#pragma unroll 2
  for (int k0 = 0; k0 < E_DIM; k0 += 32) {
    bf16x16 a  = load_bf16x16_split(a_row + k0 + a_off, a_row + k0 + 16 + a_off);
    bf16x16 b0 = load_bf16x16_split(b0_row + k0 + a_off, b0_row + k0 + 16 + a_off);
    bf16x16 b1 = load_bf16x16_split(b1_row + k0 + a_off, b1_row + k0 + 16 + a_off);
    acc0 = __builtin_amdgcn_wmma_f32_16x16x32_bf16(false, a, false, b0,
                                                   (short)0, acc0, false, false);
    acc1 = __builtin_amdgcn_wmma_f32_16x16x32_bf16(false, a, false, b1,
                                                   (short)0, acc1, false, false);
  }

  // C/D layout: lane 0-15 -> N=lane, M=r; lane 16-31 -> M=8+r.
  const int mBase = laneHi * 8;
#pragma unroll
  for (int r = 0; r < 8; ++r) {
    const int m  = row0 + mBase + r;
    const int c0 = col0 + laneLo;
    const int c1 = c0 + 16;
    Gtab[(size_t)m * G_DIM + c0] = acc0[r] + b_ih[c0] + b_hh[c0];
    Gtab[(size_t)m * G_DIM + c1] = acc1[r] + b_ih[c1] + b_hh[c1];
  }
}

// ---------------------------------------------------------------------------
// One LSTM timestep, fused: gates = h @ W_hh^T (WMMA bf16) + G[x_t[b]] lookup,
// then pointwise c/h update. Each workgroup owns a 32(b) x 32(h) tile and
// computes all four gate blocks for it so the nonlinearity stays on-chip.
// Waves: (gate 0..3) x (m-half 0..1); each wave: 16x32 tile, 2 accumulators.
__global__ __launch_bounds__(256) void lstm_step_kernel(
    const unsigned short* __restrict__ h_bf,     // [B][H] bf16 (prev step)
    const unsigned short* __restrict__ whh_bf,   // [4H][H] bf16
    const float* __restrict__ Gtab,              // [V][4H]
    const int* __restrict__ x_t,                 // [B]
    float* __restrict__ c_state,                 // [B][H]
    float* __restrict__ h_state,                 // [B][H]
    unsigned short* __restrict__ h_bf_out,       // [B][H] bf16 (next step)
    float* __restrict__ outs,                    // [B][T][H]
    int t) {
  __shared__ float g_lds[4][32][33];             // +1 col pad vs bank conflicts

  const int lane   = threadIdx.x & 31;
  const int wave   = threadIdx.x >> 5;
  const int gate   = wave & 3;                   // 0:i 1:f 2:g 3:o
  const int mhalf  = wave >> 2;                  // rows 0-15 / 16-31 of tile
  const int bRow0  = blockIdx.y * 32;
  const int hCol0  = blockIdx.x * 32;
  const int laneLo = lane & 15;
  const int laneHi = lane >> 4;

  // A fragment: h rows (M = batch), 16x32 bf16 layout:
  //   lanes 0-15: K[0..7] then K[16..23]; lanes 16-31: K[8..15] then K[24..31].
  const unsigned short* a_row =
      h_bf + (size_t)(bRow0 + mhalf * 16 + laneLo) * H_DIM;
  const int a_off = laneHi * 8;
  // B fragment: W_hh row g supplies column g of W_hh^T (contiguous in k):
  //   lanes 0-15: K[0..7] then K[16..23]; lanes 16-31: K[8..15] then K[24..31].
  const int col0 = gate * H_DIM + hCol0 + laneLo;
  const unsigned short* b0_row = whh_bf + (size_t)col0 * H_DIM;
  const unsigned short* b1_row = b0_row + (size_t)16 * H_DIM;

  f32x8 acc0 = {};
  f32x8 acc1 = {};
#pragma unroll 2
  for (int k0 = 0; k0 < H_DIM; k0 += 32) {
    bf16x16 a  = load_bf16x16_split(a_row + k0 + a_off, a_row + k0 + 16 + a_off);
    bf16x16 b0 = load_bf16x16_split(b0_row + k0 + a_off, b0_row + k0 + 16 + a_off);
    bf16x16 b1 = load_bf16x16_split(b1_row + k0 + a_off, b1_row + k0 + 16 + a_off);
    acc0 = __builtin_amdgcn_wmma_f32_16x16x32_bf16(false, a, false, b0,
                                                   (short)0, acc0, false, false);
    acc1 = __builtin_amdgcn_wmma_f32_16x16x32_bf16(false, a, false, b1,
                                                   (short)0, acc1, false, false);
  }

  // Stage accumulators to LDS: acc VGPR r holds (m = r + 8*laneHi, n = laneLo).
  const int mBase = mhalf * 16 + laneHi * 8;
#pragma unroll
  for (int r = 0; r < 8; ++r) {
    g_lds[gate][mBase + r][laneLo]      = acc0[r];
    g_lds[gate][mBase + r][laneLo + 16] = acc1[r];
  }
  __syncthreads();

  // Pointwise LSTM cell update: 1024 elements, 4 per thread.
  for (int e = threadIdx.x; e < 32 * 32; e += 256) {
    const int bi = e >> 5;
    const int hi = e & 31;
    const int b  = bRow0 + bi;
    const int h  = hCol0 + hi;
    const int v  = x_t[b];
    const float* gr = Gtab + (size_t)v * G_DIM + h;  // input-side gates + bias
    const float ig = g_lds[0][bi][hi] + gr[0 * H_DIM];
    const float fg = g_lds[1][bi][hi] + gr[1 * H_DIM];
    const float gg = g_lds[2][bi][hi] + gr[2 * H_DIM];
    const float og = g_lds[3][bi][hi] + gr[3 * H_DIM];

    const size_t sidx = (size_t)b * H_DIM + h;
    const float c_old = c_state[sidx];
    const float c_new = sigmoidf_fast(fg) * c_old + sigmoidf_fast(ig) * tanhf_fast(gg);
    const float h_new = sigmoidf_fast(og) * tanhf_fast(c_new);

    c_state[sidx]  = c_new;
    h_state[sidx]  = h_new;                     // final-h region of d_out
    h_bf_out[sidx] = f32_to_bf16_rne(h_new);    // feeds next step's WMMA
    outs[((size_t)b * T_DIM + t) * H_DIM + h] = h_new;
  }
}

// ---------------------------------------------------------------------------
extern "C" void kernel_launch(void* const* d_in, const int* in_sizes, int n_in,
                              void* d_out, int out_size, void* d_ws, size_t ws_size,
                              hipStream_t stream) {
  (void)in_sizes; (void)n_in; (void)out_size; (void)ws_size;

  const int*   x_seq = (const int*)d_in[0];    // [T][B]
  const float* emb   = (const float*)d_in[1];  // [V][E]
  const float* W_ih  = (const float*)d_in[2];  // [4H][E]
  const float* W_hh  = (const float*)d_in[3];  // [4H][H]
  const float* b_ih  = (const float*)d_in[4];  // [4H]
  const float* b_hh  = (const float*)d_in[5];  // [4H]

  // d_out = outs [B,T,H] ++ h [B,H] ++ c [B,H]; h/c double as running state.
  float* outs    = (float*)d_out;
  float* h_state = outs + (size_t)B_DIM * T_DIM * H_DIM;
  float* c_state = h_state + (size_t)B_DIM * H_DIM;

  // Workspace carve-up (~4.8 MB total).
  char* ws = (char*)d_ws;
  unsigned short* whh_bf = (unsigned short*)ws; ws += (size_t)G_DIM * H_DIM * 2;
  unsigned short* wih_bf = (unsigned short*)ws; ws += (size_t)G_DIM * E_DIM * 2;
  unsigned short* emb_bf = (unsigned short*)ws; ws += (size_t)V_DIM * E_DIM * 2;
  float*          Gtab   = (float*)ws;          ws += (size_t)V_DIM * G_DIM * 4;
  unsigned short* h_bf_a = (unsigned short*)ws; ws += (size_t)B_DIM * H_DIM * 2;
  unsigned short* h_bf_b = (unsigned short*)ws;

  cvt_f32_bf16_kernel<<<(G_DIM * H_DIM + 255) / 256, 256, 0, stream>>>(
      W_hh, whh_bf, G_DIM * H_DIM);
  cvt_f32_bf16_kernel<<<(G_DIM * E_DIM + 255) / 256, 256, 0, stream>>>(
      W_ih, wih_bf, G_DIM * E_DIM);
  cvt_f32_bf16_kernel<<<(V_DIM * E_DIM + 255) / 256, 256, 0, stream>>>(
      emb, emb_bf, V_DIM * E_DIM);
  init_state_kernel<<<(B_DIM * H_DIM + 255) / 256, 256, 0, stream>>>(
      c_state, h_state, h_bf_a, B_DIM * H_DIM);

  // Fold embedding + input projection + both biases into G[V][4H].
  g_table_kernel<<<dim3(G_DIM / 256, V_DIM / 16), 256, 0, stream>>>(
      emb_bf, wih_bf, b_ih, b_hh, Gtab);

  // Sequential scan; bf16 hidden state ping-pongs to avoid cross-block races.
  for (int t = 0; t < T_DIM; ++t) {
    const unsigned short* h_in  = (t & 1) ? h_bf_b : h_bf_a;
    unsigned short*       h_out = (t & 1) ? h_bf_a : h_bf_b;
    lstm_step_kernel<<<dim3(H_DIM / 32, B_DIM / 32), 256, 0, stream>>>(
        h_in, whh_bf, Gtab, x_seq + (size_t)t * B_DIM,
        c_state, h_state, h_out, outs, t);
  }
}
